// _NonLocalNd_bn_42030549958734
// MI455X (gfx1250) — matrix operand order repaired and checked by reading, not compiled, not run on hardware
//
#include <hip/hip_runtime.h>
#include <hip/hip_fp16.h>

// ---------------------------------------------------------------------------
// NonLocal attention block for MI455X (gfx1250, wave32, WMMA).
// All GEMMs on v_wmma_f32_16x16x32_f16 (f16 in, f32 accumulate; f16 chosen
// over bf16: this workload's tensors have tiny dynamic range, f16's 10
// mantissa bits win, and f16 scores halve LDS so 2 WGs co-reside per WGP).
// Operands pre-swizzled into wave32 WMMA fragment layout in global memory so
// every fragment is one 32B vector load per lane.
// Projection GEMM: 64-wide k-steps, ping-pong LDS double buffering.
// Attention: full 64x1024 f16 score block in LDS (~147KB -> 2 WGs/WGP =
// 2 waves/SIMD for latency hiding) -> exact single-pass softmax.
// ---------------------------------------------------------------------------

typedef __attribute__((ext_vector_type(16))) _Float16 f16x16;
typedef __attribute__((ext_vector_type(8)))  _Float16 f16x8;
typedef __attribute__((ext_vector_type(4)))  _Float16 f16x4;
typedef __attribute__((ext_vector_type(8)))  float    f32x8;
typedef __attribute__((ext_vector_type(4)))  float    f32x4;

#define B_  8
#define C_  512
#define H_  64
#define W_  64
#define N_  4096   // H*W
#define M_  1024   // (H/2)*(W/2)
#define P_  128    // PLANES
#define INV_SCALE 1.7677669529663689f   // 1 / (sqrt(128) * 0.05)
#define SROWH 1040 // padded f16 score-row stride (16B-aligned e-runs)

// ---------------- fragment-layout indexers (wave32 WMMA, CDNA5 ISA) --------
// A 16x32 f16: lane = row + 16*(k-subgroup), elems = 2 runs of 8 k's.
// B 32x16 f16: lane = col + 16*(k-subgroup), elems = 2 runs of 8 k's (same as A).

// q as A-matrix of S = q^T k : row = n, K = p.
__device__ __forceinline__ size_t q_index(int b, int n, int p) {
    int nblk = n >> 6, rg = (n >> 4) & 3, rl = n & 15;
    int ks = p >> 5, kk = p & 31;
    int lane = rl + (((kk >> 3) & 1) << 4);
    int e = (kk & 7) + ((kk >> 4) << 3);
    return ((((size_t)((b * 64 + nblk) * 4 + rg)) * 4 + ks) * 32 + lane) * 16 + e;
}
// k as B-matrix of S : col = m, K = p.
__device__ __forceinline__ size_t k_index(int b, int m, int p) {
    int mblk = m >> 6, t = (m >> 4) & 3, cl = m & 15;
    int ks = p >> 5, kk = p & 31;
    int lane = cl + (((kk >> 3) & 1) << 4);
    int e = (kk & 7) + ((kk >> 4) << 3);
    return ((((size_t)((b * 16 + mblk) * 4 + t)) * 4 + ks) * 32 + lane) * 16 + e;
}
// v as B-matrix of O = P v^T : col = c, K = m.
__device__ __forceinline__ size_t v_index(int b, int c, int m) {
    int cblk = c >> 6, t = (c >> 4) & 3, cl = c & 15;
    int ms = m >> 5, mm = m & 31;
    int lane = cl + (((mm >> 3) & 1) << 4);
    int e = (mm & 7) + ((mm >> 4) << 3);
    return ((((size_t)((b * 8 + cblk) * 4 + t)) * 32 + ms) * 32 + lane) * 16 + e;
}

// ---------------- Stage 1: 2x2 maxpool ------------------------------------

__global__ void __launch_bounds__(256)
maxpool_kernel(const float* __restrict__ x, float* __restrict__ xd) {
    int i = blockIdx.x * 256 + threadIdx.x;          // over B*C*M
    if (i >= B_ * C_ * M_) return;
    int mx = i & 31, my = (i >> 5) & 31, bc = i >> 10;
    const float* p = x + ((size_t)bc * H_ + 2 * my) * W_ + 2 * mx;
    xd[i] = fmaxf(fmaxf(p[0], p[1]), fmaxf(p[W_], p[W_ + 1]));
}

// ---------------- Stage 2: projection GEMM (WMMA f16) ----------------------
// C[b][row][col] = sum_k Wt[row][k] * X[b][k][col] (+ bias[row])
// 128 threads (4 waves), 64x64 tile, 64-wide k-step, ping-pong LDS buffers.
// Fragment-layout LDS: A [rowgrp(4)][ks(2)][lane][16], B [t(4)][ks(2)][lane][16].

__device__ __forceinline__ void stage64(const float* __restrict__ Wt,
                                        const float* __restrict__ Xb,
                                        _Float16* __restrict__ sA,
                                        _Float16* __restrict__ sB,
                                        int row0, int col0, int k0,
                                        int K, int Ncols, int tid) {
    // A tile: 64 rows x 64 k, f32x4 reads -> f16x4 runs
    for (int i = tid; i < 1024; i += 128) {
        int lin = i << 2;
        int r = lin >> 6, kk0 = lin & 63;
        f32x4 w4 = *(const f32x4*)(Wt + (size_t)(row0 + r) * K + k0 + kk0);
        int ks = kk0 >> 5, kk5 = kk0 & 31;
        int hi = ((kk5 >> 3) & 1) << 4;
        int e0 = (kk5 & 7) + ((kk5 >> 4) << 3);
        int slot = ((r >> 4) * 2 + ks) * 32 + (r & 15) + hi;
        f16x4 v4 = { (_Float16)w4[0], (_Float16)w4[1], (_Float16)w4[2], (_Float16)w4[3] };
        *(f16x4*)(sA + slot * 16 + e0) = v4;
    }
    // B tile: 64 k x 64 c, f32x4 reads
    for (int i = tid; i < 1024; i += 128) {
        int lin = i << 2;
        int kk = lin >> 6, cc = lin & 63;
        f32x4 x4 = *(const f32x4*)(Xb + (size_t)(k0 + kk) * Ncols + col0 + cc);
        int ks = kk >> 5, kk5 = kk & 31;
        int lb = (cc & 15) + (((kk5 >> 3) & 1) << 4);
        int e = (kk5 & 7) + ((kk5 >> 4) << 3);
        int base = ((cc >> 4) * 2 + ks) * 32 + lb;
        sB[(base + 0) * 16 + e] = (_Float16)x4[0];
        sB[(base + 1) * 16 + e] = (_Float16)x4[1];
        sB[(base + 2) * 16 + e] = (_Float16)x4[2];
        sB[(base + 3) * 16 + e] = (_Float16)x4[3];
    }
}

__global__ void __launch_bounds__(128)
gemm_proj(const float* __restrict__ Wt, const float* __restrict__ X,
          const float* __restrict__ bias, float* __restrict__ Cout,
          int Mrows, int K, int Ncols) {
    __shared__ alignas(32) _Float16 sA[2][4096];
    __shared__ alignas(32) _Float16 sB[2][4096];
    int b    = blockIdx.z;
    int row0 = blockIdx.y * 64;
    int col0 = blockIdx.x * 64;
    int tid  = threadIdx.x, wave = tid >> 5, lane = tid & 31;
    const float* Xb = X + (size_t)b * K * Ncols;
    float* Cb = Cout + (size_t)b * Mrows * Ncols;

    f32x8 acc[4] = {};
    stage64(Wt, Xb, sA[0], sB[0], row0, col0, 0, K, Ncols, tid);
    int nkt = K / 64;
    for (int kt = 0; kt < nkt; ++kt) {
        __syncthreads();   // staged buffer ready; previous compute done
        if (kt + 1 < nkt)
            stage64(Wt, Xb, sA[(kt + 1) & 1], sB[(kt + 1) & 1],
                    row0, col0, (kt + 1) * 64, K, Ncols, tid);
        const _Float16* pa = sA[kt & 1];
        const _Float16* pb = sB[kt & 1];
#pragma unroll
        for (int ks = 0; ks < 2; ++ks) {
            f16x16 a = *(const f16x16*)(pa + ((wave * 2 + ks) * 32 + lane) * 16);
            f16x16 bb[4];
#pragma unroll
            for (int t = 0; t < 4; ++t)
                bb[t] = *(const f16x16*)(pb + ((t * 2 + ks) * 32 + lane) * 16);
#pragma unroll
            for (int t = 0; t < 4; ++t)
                acc[t] = __builtin_amdgcn_wmma_f32_16x16x32_f16(
                    false, a, false, bb[t], (short)0, acc[t], false, false);
        }
    }
    int mr8 = (lane >> 4) << 3, cN = lane & 15;
#pragma unroll
    for (int t = 0; t < 4; ++t)
#pragma unroll
        for (int e = 0; e < 8; ++e) {
            int row = row0 + wave * 16 + e + mr8;
            int col = col0 + t * 16 + cN;
            float v = acc[t][e] + (bias ? bias[row] : 0.0f);
            Cb[(size_t)row * Ncols + col] = v;
        }
}

// ---------------- Stage 3: whitening -> fragment-layout f16 -----------------

__global__ void __launch_bounds__(256)
whiten_q_kernel(const float* __restrict__ in, _Float16* __restrict__ outq) {
    __shared__ float red[256];
    int blk = blockIdx.x;                 // b*128 + p
    int b = blk >> 7, p = blk & 127;
    const float* row = in + (size_t)blk * N_;
    int tid = threadIdx.x;
    float s = 0.f;
    for (int i = tid; i < N_; i += 256) s += row[i];
    red[tid] = s; __syncthreads();
    for (int st = 128; st > 0; st >>= 1) {
        if (tid < st) red[tid] += red[tid + st];
        __syncthreads();
    }
    float mean = red[0] / (float)N_;
    for (int n = tid; n < N_; n += 256)
        outq[q_index(b, n, p)] = (_Float16)(row[n] - mean);
}

__global__ void __launch_bounds__(256)
whiten_k_kernel(const float* __restrict__ in, _Float16* __restrict__ outk) {
    __shared__ float red[256];
    int blk = blockIdx.x;                 // b*128 + p
    int b = blk >> 7, p = blk & 127;
    const float* row = in + (size_t)blk * M_;
    int tid = threadIdx.x;
    float s = 0.f;
    for (int i = tid; i < M_; i += 256) s += row[i];
    red[tid] = s; __syncthreads();
    for (int st = 128; st > 0; st >>= 1) {
        if (tid < st) red[tid] += red[tid + st];
        __syncthreads();
    }
    float mean = red[0] / (float)M_;
    for (int m = tid; m < M_; m += 256)
        outk[k_index(b, m, p)] = (_Float16)(row[m] - mean);
}

__global__ void __launch_bounds__(256)
cast_v_kernel(const float* __restrict__ vf32, _Float16* __restrict__ vfr) {
    int i = blockIdx.x * 256 + threadIdx.x;     // over B*C*M
    if (i >= B_ * C_ * M_) return;
    int m = i & (M_ - 1), c = (i >> 10) & (C_ - 1), b = i >> 19;
    vfr[v_index(b, c, m)] = (_Float16)vf32[i];
}

// ---------------- Stage 4: mask softmax + global-context branch ------------

__global__ void __launch_bounds__(256)
mask_gc_kernel(const float* __restrict__ xd, const float* __restrict__ wm,
               const float* __restrict__ bm, const float* __restrict__ vf,
               float* __restrict__ gc) {
    __shared__ float sm[M_];
    __shared__ float red[256];
    int b = blockIdx.x, tid = threadIdx.x;
    const float* xdb = xd + (size_t)b * C_ * M_;
    for (int m = tid; m < M_; m += 256) {
        float s = bm[0];
        for (int c = 0; c < C_; ++c) s += wm[c] * xdb[(size_t)c * M_ + m];
        sm[m] = s;
    }
    __syncthreads();
    float mx = -3.0e38f;
    for (int m = tid; m < M_; m += 256) mx = fmaxf(mx, sm[m]);
    red[tid] = mx; __syncthreads();
    for (int st = 128; st > 0; st >>= 1) {
        if (tid < st) red[tid] = fmaxf(red[tid], red[tid + st]);
        __syncthreads();
    }
    mx = red[0]; __syncthreads();
    float sum = 0.f;
    for (int m = tid; m < M_; m += 256) {
        float e = __expf(sm[m] - mx);
        sm[m] = e;
        sum += e;
    }
    red[tid] = sum; __syncthreads();
    for (int st = 128; st > 0; st >>= 1) {
        if (tid < st) red[tid] += red[tid + st];
        __syncthreads();
    }
    float inv = 1.0f / red[0];
    const float* vb = vf + (size_t)b * C_ * M_;
    for (int c = tid; c < C_; c += 256) {
        float s = 0.f;
        for (int m = 0; m < M_; ++m) s += vb[(size_t)c * M_ + m] * sm[m];
        gc[b * C_ + c] = s * inv;
    }
}

// ---------------- Stage 5: fused attention ---------------------------------
// Grid (N/64, B). 128 threads = 4 waves, WG owns 64 rows of N.
// Fragments come straight from global (pre-swizzled, L2-resident) -> no LDS
// staging, no barriers inside GEMM loops. Scores (f16) in LDS, exact softmax.
// ~147KB LDS -> 2 WGs/WGP (2 waves/SIMD) for latency hiding.
// Output bounced through small LDS transpose tile for coalesced f32x4 stores.

__global__ void __launch_bounds__(128)
attn_kernel(const _Float16* __restrict__ qf, const _Float16* __restrict__ kf,
            const _Float16* __restrict__ vfr, const float* __restrict__ gc,
            const float* __restrict__ x, const float* __restrict__ gamma,
            float* __restrict__ out) {
    extern __shared__ char smem[];
    _Float16* sS = (_Float16*)smem;                             // [64][SROWH]
    float*    sL = (float*)(smem + (size_t)64 * SROWH * 2);     // [64]
    float*    sX = (float*)(smem + (size_t)64 * SROWH * 2 + 256); // [64][68] c-major

    int b = blockIdx.y, nblk = blockIdx.x;
    int n0 = nblk * 64;
    int tid = threadIdx.x, wave = tid >> 5, lane = tid & 31;
    int mr8 = (lane >> 4) << 3, cN = lane & 15;

    // Preload this wave's 4 Q fragments (rows [16*wave,16*wave+16), K=0..127)
    const _Float16* qbase = qf + (((size_t)((b * 64 + nblk) * 4 + wave)) * 4) * 512
                               + (size_t)lane * 16;
    f16x16 qfrag[4];
#pragma unroll
    for (int ks = 0; ks < 4; ++ks)
        qfrag[ks] = *(const f16x16*)(qbase + (size_t)ks * 512);

    // ---- Pass 1: S = (q^T k) * inv_scale ----
    for (int mb = 0; mb < M_ / 64; ++mb) {
        f32x8 acc[4] = {};
        const _Float16* kbase = kf + ((size_t)(b * 16 + mb) * 16) * 512 + (size_t)lane * 16;
#pragma unroll
        for (int ks = 0; ks < 4; ++ks) {
            f16x16 bb[4];
#pragma unroll
            for (int t = 0; t < 4; ++t)
                bb[t] = *(const f16x16*)(kbase + (size_t)(t * 4 + ks) * 512);
#pragma unroll
            for (int t = 0; t < 4; ++t)
                acc[t] = __builtin_amdgcn_wmma_f32_16x16x32_f16(
                    false, qfrag[ks], false, bb[t], (short)0, acc[t], false, false);
        }
#pragma unroll
        for (int t = 0; t < 4; ++t)
#pragma unroll
            for (int e = 0; e < 8; ++e) {
                int row = wave * 16 + e + mr8;
                sS[row * SROWH + mb * 64 + t * 16 + cN] =
                    (_Float16)(acc[t][e] * INV_SCALE);
            }
    }
    __syncthreads();

    // ---- Exact softmax per row (thread r owns row r) ----
    if (tid < 64) {
        _Float16* r = sS + (size_t)tid * SROWH;
        float mx = -3.0e38f;
        for (int m = 0; m < M_; ++m) mx = fmaxf(mx, (float)r[m]);
        float sum = 0.f;
        for (int m = 0; m < M_; ++m) {
            float e = __expf((float)r[m] - mx);
            r[m] = (_Float16)e;
            sum += e;
        }
        sL[tid] = sum;
    }
    __syncthreads();

    float g = gamma[0];
    // A-fragment source row for pass 2 (lane-private): row = wave*16 + (lane&15)
    const _Float16* arow = sS + (size_t)(wave * 16 + cN) * SROWH + ((lane & 16) ? 8 : 0);

    // ---- Pass 2: O = P @ v^T (K = 1024), fused epilogue ----
    for (int cb = 0; cb < C_ / 64; ++cb) {
        f32x8 acc[4] = {};
        const _Float16* vbase = vfr + ((size_t)(b * 8 + cb) * 4) * 32 * 512
                                    + (size_t)lane * 16;
        for (int ms = 0; ms < 32; ++ms) {
            const _Float16* pr = arow + ms * 32;
            f16x8 a0 = *(const f16x8*)(pr);        // e0..7  (16B, ds_load_b128)
            f16x8 a1 = *(const f16x8*)(pr + 16);   // e8..15 (16B, ds_load_b128)
            f16x16 a = __builtin_shufflevector(a0, a1,
                        0, 1, 2, 3, 4, 5, 6, 7, 8, 9, 10, 11, 12, 13, 14, 15);
            f16x16 bb[4];
#pragma unroll
            for (int t = 0; t < 4; ++t)
                bb[t] = *(const f16x16*)(vbase + (size_t)(t * 32 + ms) * 512);
#pragma unroll
            for (int t = 0; t < 4; ++t)
                acc[t] = __builtin_amdgcn_wmma_f32_16x16x32_f16(
                    false, a, false, bb[t], (short)0, acc[t], false, false);
        }
        // transpose through LDS for coalesced [B,C,N] stores
        __syncthreads();
#pragma unroll
        for (int t = 0; t < 4; ++t)
#pragma unroll
            for (int e = 0; e < 8; ++e) {
                int rloc = wave * 16 + e + mr8;
                sX[(t * 16 + cN) * 68 + rloc] = acc[t][e];
            }
        __syncthreads();
        int c0 = cb * 64;
        for (int i = tid; i < 64 * 16; i += 128) {
            int cL = i >> 4, nL = (i & 15) << 2;
            int c = c0 + cL;
            size_t idx = ((size_t)b * C_ + c) * N_ + n0 + nL;
            f32x4 o   = *(const f32x4*)(sX + cL * 68 + nL);
            f32x4 xin = *(const f32x4*)(x + idx);
            float gcv = gc[b * C_ + c];
            f32x4 r;
#pragma unroll
            for (int j = 0; j < 4; ++j) r[j] = g * (o[j] / sL[nL + j]) + gcv + xin[j];
            *(f32x4*)(out + idx) = r;
        }
    }
}

// ---------------------------------------------------------------------------

extern "C" void kernel_launch(void* const* d_in, const int* in_sizes, int n_in,
                              void* d_out, int out_size, void* d_ws, size_t ws_size,
                              hipStream_t stream) {
    (void)in_sizes; (void)n_in; (void)out_size; (void)ws_size;
    const float* x     = (const float*)d_in[0];
    const float* wq    = (const float*)d_in[1];
    const float* bq    = (const float*)d_in[2];
    const float* wk    = (const float*)d_in[3];
    const float* bk    = (const float*)d_in[4];
    const float* wv    = (const float*)d_in[5];
    const float* wm    = (const float*)d_in[6];
    const float* bm    = (const float*)d_in[7];
    const float* gamma = (const float*)d_in[8];
    float* out = (float*)d_out;

    char* ws = (char*)d_ws;
    size_t off = 0;
    auto take = [&](size_t bytes) -> char* {
        char* p = ws + off;
        off = (off + bytes + 255) & ~(size_t)255;
        return p;
    };
    float*     xd  = (float*)take((size_t)B_ * C_ * M_ * 4);      // pooled input
    float*     qf  = (float*)take((size_t)B_ * P_ * N_ * 4);      // q f32 (pre-whiten)
    float*     kf  = (float*)take((size_t)B_ * P_ * M_ * 4);      // k f32 (pre-whiten)
    float*     vf  = (float*)take((size_t)B_ * C_ * M_ * 4);      // v f32 (gc branch)
    _Float16*  qfr = (_Float16*)take((size_t)B_ * P_ * N_ * 2);   // q frag-layout f16
    _Float16*  kfr = (_Float16*)take((size_t)B_ * P_ * M_ * 2);   // k frag-layout f16
    _Float16*  vfr = (_Float16*)take((size_t)B_ * C_ * M_ * 2);   // v frag-layout f16
    float*     gc  = (float*)take((size_t)B_ * C_ * 4);           // global-context

    maxpool_kernel<<<(B_ * C_ * M_) / 256, 256, 0, stream>>>(x, xd);

    gemm_proj<<<dim3(N_ / 64, P_ / 64, B_), 128, 0, stream>>>(wq, x,  bq,      qf, P_, C_, N_);
    gemm_proj<<<dim3(M_ / 64, P_ / 64, B_), 128, 0, stream>>>(wk, xd, bk,      kf, P_, C_, M_);
    gemm_proj<<<dim3(M_ / 64, C_ / 64, B_), 128, 0, stream>>>(wv, xd, nullptr, vf, C_, C_, M_);

    whiten_q_kernel<<<B_ * P_, 256, 0, stream>>>(qf, qfr);
    whiten_k_kernel<<<B_ * P_, 256, 0, stream>>>(kf, kfr);
    cast_v_kernel<<<(B_ * C_ * M_) / 256, 256, 0, stream>>>(vf, vfr);

    mask_gc_kernel<<<B_, 256, 0, stream>>>(xd, wm, bm, vf, gc);

    // f16 scores: 64*1040*2 + 256 + 64*68*4 ~= 147KB -> 2 WGs per WGP (320KB)
    size_t shmem = (size_t)64 * SROWH * 2 + 256 + (size_t)64 * 68 * 4;
    hipFuncSetAttribute((const void*)attn_kernel,
                        hipFuncAttributeMaxDynamicSharedMemorySize, (int)shmem);
    attn_kernel<<<dim3(N_ / 64, B_), 128, shmem, stream>>>(qfr, kfr, vfr, gc, x, gamma, out);
}
